// LSTMDecoder_9079560863839
// MI455X (gfx1250) — matrix operand order repaired and checked by reading, not compiled, not run on hardware
//
#include <hip/hip_runtime.h>
#include <hip/hip_bf16.h>
#include <stdint.h>

// ---------------------------------------------------------------------------
// CDNA5 fused hierarchical-LSTM decoder.
//   B=65536, F=256.  All GEMMs via v_wmma_f32_16x16x32_bf16 (wave32).
//   Compute-bound (AI ~1800 FLOP/byte when fused) -> bf16 WMMA, full fusion:
//   per-32-row-tile state (x, h1, c1, h2, c2, h3_in, h3) kept in LDS,
//   bf16 weights (~4MB) transposed once into d_ws and streamed from L2.
// ---------------------------------------------------------------------------

typedef __attribute__((ext_vector_type(16))) __bf16 v16bf;
typedef __attribute__((ext_vector_type(8)))  __bf16 v8bf;
typedef __attribute__((ext_vector_type(4)))  __bf16 v4bf;
typedef __attribute__((ext_vector_type(8)))  float  v8f;

#define FDIM   256
#define TILE_M 32
#define BATCH  65536
#define NC1    103
#define NC2    119
#define NC3    12
#define NOPAD  128   // head output dims padded to 128 (multiple of 16)

static __device__ __forceinline__ v8f zero8() {
  v8f z = {0.f,0.f,0.f,0.f,0.f,0.f,0.f,0.f};
  return z;
}
static __device__ __forceinline__ float sigf(float x) {
  return 1.0f / (1.0f + __expf(-x));
}
static __device__ __forceinline__ v8f wmma_bf16(v16bf a, v16bf b, v8f c) {
  // 8-arg form: (neg_a, A, neg_b, B, c_mod, C, reuse_a, reuse_b)
  return __builtin_amdgcn_wmma_f32_16x16x32_bf16(false, a, false, b, (short)0, c,
                                                 false, false);
}

// A-fragment (16x32 bf16) from an LDS tile laid out [rows][256] bf16.
// Lane L holds row M=L%16; half hf=L/16 selects 16 consecutive K
// {hf*16..hf*16+15} (matches B fragment)  -> one 32B ds load.
static __device__ __forceinline__ v16bf ldsA(const __bf16* tile, int rowBase,
                                             int kk, int mA, int hf) {
  const __bf16* p = tile + (rowBase + mA) * FDIM + kk + hf * 16;
  return *(const v16bf*)p;
}

// B-fragment (32x16 bf16) from a transposed weight [out][256] bf16 in global.
// Lane L holds column n=L%16 with 16 consecutive K starting at kk+hf*16
// -> one contiguous 32B load per lane (weights are L2-resident).
static __device__ __forceinline__ v16bf gmemB(const __bf16* wT, int col,
                                              int kk, int hf) {
  return *(const v16bf*)(wT + (size_t)col * FDIM + kk + hf * 16);
}

// ---------------------------------------------------------------------------
// One LSTM cell over the tile: z = x@Wk (+ hprev@Wr) + b; gate math in regs.
// hprevb == nullptr  -> skip the recurrent GEMM (zero initial state).
// cinf   == nullptr  -> c_in = 0.
// ---------------------------------------------------------------------------
static __device__ __forceinline__ void lstm_phase(
    const __bf16* xb, const __bf16* hprevb, const float* cinf,
    const __bf16* wkT, const __bf16* wrT, const float* bias,
    __bf16* hOutB, float* cOutF, __bf16* cOutB,
    int rowBase, int mA, int hf, int nw)
{
  for (int pass = 0; pass < 4; ++pass) {
    const int nn = pass * 64 + nw * 16 + mA;   // hidden column owned by lane
    v8f acc[4] = {zero8(), zero8(), zero8(), zero8()};

    for (int kk = 0; kk < FDIM; kk += 32) {
      v16bf a = ldsA(xb, rowBase, kk, mA, hf);
#pragma unroll
      for (int q = 0; q < 4; ++q) {
        v16bf b = gmemB(wkT, q * FDIM + nn, kk, hf);
        acc[q] = wmma_bf16(a, b, acc[q]);
      }
    }
    if (hprevb) {
      for (int kk = 0; kk < FDIM; kk += 32) {
        v16bf a = ldsA(hprevb, rowBase, kk, mA, hf);
#pragma unroll
        for (int q = 0; q < 4; ++q) {
          v16bf b = gmemB(wrT, q * FDIM + nn, kk, hf);
          acc[q] = wmma_bf16(a, b, acc[q]);
        }
      }
    }

    const float bI = bias[0 * FDIM + nn];
    const float bF = bias[1 * FDIM + nn];
    const float bG = bias[2 * FDIM + nn];
    const float bO = bias[3 * FDIM + nn];
#pragma unroll
    for (int r = 0; r < 8; ++r) {
      const int m = rowBase + r + 8 * hf;      // C/D layout: M = r + 8*(lane/16)
      const float iv = sigf(acc[0][r] + bI);
      const float fv = sigf(acc[1][r] + bF);
      const float gv = tanhf(acc[2][r] + bG);
      const float ov = sigf(acc[3][r] + bO);
      const float cin = cinf ? cinf[m * FDIM + nn] : 0.0f;
      const float cn = fv * cin + iv * gv;
      const float hn = ov * tanhf(cn);
      hOutB[m * FDIM + nn] = (__bf16)hn;
      if (cOutF) cOutF[m * FDIM + nn] = cn;
      if (cOutB) cOutB[m * FDIM + nn] = (__bf16)cn;
    }
  }
}

// out_a = relu(h@Wm + bm) @ Wo + bo   (Wo padded to NOPAD cols)
static __device__ __forceinline__ void head_phase(
    const __bf16* hb, const __bf16* wmT, const float* bm,
    const __bf16* woT, const float* bo, float* outBase, int dim,
    __bf16* midb, int rowBase, int row0, int mA, int hf, int nw)
{
  for (int pass = 0; pass < 4; ++pass) {
    const int nn = pass * 64 + nw * 16 + mA;
    v8f acc = zero8();
    for (int kk = 0; kk < FDIM; kk += 32) {
      v16bf a = ldsA(hb, rowBase, kk, mA, hf);
      v16bf b = gmemB(wmT, nn, kk, hf);
      acc = wmma_bf16(a, b, acc);
    }
    const float bv = bm[nn];
#pragma unroll
    for (int r = 0; r < 8; ++r) {
      const int m = rowBase + r + 8 * hf;
      const float v = acc[r] + bv;
      midb[m * FDIM + nn] = (__bf16)(v > 0.0f ? v : 0.0f);
    }
  }
  __syncthreads();

  for (int pass = 0; pass < 2; ++pass) {      // NOPAD=128 output cols
    const int nn = pass * 64 + nw * 16 + mA;
    v8f acc = zero8();
    for (int kk = 0; kk < FDIM; kk += 32) {
      v16bf a = ldsA(midb, rowBase, kk, mA, hf);
      v16bf b = gmemB(woT, nn, kk, hf);
      acc = wmma_bf16(a, b, acc);
    }
    if (nn < dim) {                            // divergence only around stores
      const float bv = bo[nn];
#pragma unroll
      for (int r = 0; r < 8; ++r) {
        const int m = rowBase + r + 8 * hf;
        outBase[(size_t)(row0 + m) * dim + nn] = acc[r] + bv;
      }
    }
  }
  __syncthreads();
}

struct FusedParams {
  const float* x;
  const __bf16* wkT1; const __bf16* wrT1;
  const __bf16* wkT2; const __bf16* wrT2;
  const __bf16* wkT3; const __bf16* wrT3;
  const __bf16* wh23T; const __bf16* wc23T;
  const __bf16* wmT1; const __bf16* wmT2; const __bf16* wmT3;
  const __bf16* woT1; const __bf16* woT2; const __bf16* woT3;
  const float* b1; const float* b2; const float* b3;
  const float* bh23; const float* bc23;
  const float* bm1; const float* bm2; const float* bm3;
  const float* bo1; const float* bo2; const float* bo3;
  float* out;
};

// LDS pool (128 KB dynamic) per 32-row tile:
//   xb 16K | h1b 16K | h2b 16K | c2b 16K | h3inb 16K | h3b 16K | c1f 32K
__global__ __launch_bounds__(256) void fused_lstm_decoder(FusedParams p)
{
  extern __shared__ char smem[];
  __bf16* xb    = (__bf16*)(smem);
  __bf16* h1b   = (__bf16*)(smem + 16384);
  __bf16* h2b   = (__bf16*)(smem + 32768);
  __bf16* c2b   = (__bf16*)(smem + 49152);
  __bf16* h3inb = (__bf16*)(smem + 65536);
  __bf16* h3b   = (__bf16*)(smem + 81920);
  float*  c1f   = (float*)(smem + 98304);   // c1, later updated in place to c3_in
  __bf16* midb  = xb;                       // x dead after LSTM3 -> reuse for heads

  const int t    = threadIdx.x;
  const int lane = t & 31;
  const int wave = t >> 5;
  const int mA   = lane & 15;
  const int hf   = lane >> 4;
  const int mw   = wave >> 2;               // 0..1: 16-row block
  const int nw   = wave & 3;                // 0..3: 16-col block per pass
  const int rowBase = mw * 16;
  const int row0 = blockIdx.x * TILE_M;

  // ---- stage x tile, f32 -> bf16 in LDS (coalesced float4 loads) ----
  {
    const int c4 = (t & 63) * 4;
    const int rw = t >> 6;                  // 0..3
    for (int j = 0; j < 8; ++j) {
      const int rr = rw + j * 4;
      float4 v = *(const float4*)(p.x + (size_t)(row0 + rr) * FDIM + c4);
      v4bf bv;
      bv[0] = (__bf16)v.x; bv[1] = (__bf16)v.y;
      bv[2] = (__bf16)v.z; bv[3] = (__bf16)v.w;
      *(v4bf*)(xb + rr * FDIM + c4) = bv;
    }
  }
  __syncthreads();

  // ---- LSTM1: zero initial state -> skip h@Wr1 GEMM entirely ----
  lstm_phase(xb, (const __bf16*)nullptr, (const float*)nullptr,
             p.wkT1, (const __bf16*)nullptr, p.b1,
             h1b, c1f, (__bf16*)nullptr, rowBase, mA, hf, nw);
  __syncthreads();

  // ---- LSTM2: state copied from task 1 (first visit) ----
  lstm_phase(xb, h1b, c1f, p.wkT2, p.wrT2, p.b2,
             h2b, (float*)nullptr, c2b, rowBase, mA, hf, nw);
  __syncthreads();

  // ---- edge 2->3: h3_in = h1 + h2@Wh23 + bh23 ; c3_in = c1 + c2@Wc23 + bc23 ----
  for (int pass = 0; pass < 4; ++pass) {
    const int nn = pass * 64 + nw * 16 + mA;
    v8f aH = zero8(), aC = zero8();
    for (int kk = 0; kk < FDIM; kk += 32) {
      v16bf ah = ldsA(h2b, rowBase, kk, mA, hf);
      v16bf bh = gmemB(p.wh23T, nn, kk, hf);
      aH = wmma_bf16(ah, bh, aH);
      v16bf ac = ldsA(c2b, rowBase, kk, mA, hf);
      v16bf bc = gmemB(p.wc23T, nn, kk, hf);
      aC = wmma_bf16(ac, bc, aC);
    }
    const float bhv = p.bh23[nn];
    const float bcv = p.bc23[nn];
#pragma unroll
    for (int r = 0; r < 8; ++r) {
      const int m = rowBase + r + 8 * hf;
      const float h1v = (float)h1b[m * FDIM + nn];
      h3inb[m * FDIM + nn] = (__bf16)(h1v + aH[r] + bhv);
      c1f[m * FDIM + nn]   = c1f[m * FDIM + nn] + aC[r] + bcv;   // becomes c3_in
    }
  }
  __syncthreads();

  // ---- LSTM3 ----
  lstm_phase(xb, h3inb, c1f, p.wkT3, p.wrT3, p.b3,
             h3b, (float*)nullptr, (__bf16*)nullptr, rowBase, mA, hf, nw);
  __syncthreads();

  // ---- heads (midb aliases xb; head_phase barriers internally) ----
  float* out1 = p.out;
  float* out2 = p.out + (size_t)BATCH * NC1;
  float* out3 = p.out + (size_t)BATCH * (NC1 + NC2);
  head_phase(h1b, p.wmT1, p.bm1, p.woT1, p.bo1, out1, NC1, midb, rowBase, row0, mA, hf, nw);
  head_phase(h2b, p.wmT2, p.bm2, p.woT2, p.bo2, out2, NC2, midb, rowBase, row0, mA, hf, nw);
  head_phase(h3b, p.wmT3, p.bm3, p.woT3, p.bo3, out3, NC3, midb, rowBase, row0, mA, hf, nw);
}

// Transpose+convert weight [256][N] f32 -> [Npad][256] bf16 (zero-pad cols>=N).
__global__ __launch_bounds__(256) void prep_wT_bf16(const float* __restrict__ src,
                                                    __bf16* __restrict__ dst,
                                                    int N, int Npad)
{
  const int idx = blockIdx.x * 256 + threadIdx.x;
  if (idx >= Npad * FDIM) return;
  const int n = idx >> 8;
  const int k = idx & (FDIM - 1);
  const float v = (n < N) ? src[(size_t)k * N + n] : 0.0f;
  dst[idx] = (__bf16)v;
}

extern "C" void kernel_launch(void* const* d_in, const int* in_sizes, int n_in,
                              void* d_out, int out_size, void* d_ws, size_t ws_size,
                              hipStream_t stream)
{
  const float* x     = (const float*)d_in[0];
  const float* Wk[3] = {(const float*)d_in[1],  (const float*)d_in[4],  (const float*)d_in[7]};
  const float* Wr[3] = {(const float*)d_in[2],  (const float*)d_in[5],  (const float*)d_in[8]};
  const float* b [3] = {(const float*)d_in[3],  (const float*)d_in[6],  (const float*)d_in[9]};
  const float* Wh23  = (const float*)d_in[10];  const float* bh23 = (const float*)d_in[11];
  const float* Wc23  = (const float*)d_in[12];  const float* bc23 = (const float*)d_in[13];
  const float* Wm[3] = {(const float*)d_in[14], (const float*)d_in[18], (const float*)d_in[22]};
  const float* bm[3] = {(const float*)d_in[15], (const float*)d_in[19], (const float*)d_in[23]};
  const float* Wo[3] = {(const float*)d_in[16], (const float*)d_in[20], (const float*)d_in[24]};
  const float* bo[3] = {(const float*)d_in[17], (const float*)d_in[21], (const float*)d_in[25]};

  // bf16 workspace layout (element offsets); total ~3.8 MB -> L2-resident.
  __bf16* ws = (__bf16*)d_ws;
  __bf16* wkT[3] = {ws + 0,       ws + 524288,  ws + 1048576};
  __bf16* wrT[3] = {ws + 262144,  ws + 786432,  ws + 1310720};
  __bf16* wh23T  = ws + 1572864;
  __bf16* wc23T  = ws + 1638400;
  __bf16* wmT[3] = {ws + 1703936, ws + 1769472, ws + 1835008};
  __bf16* woT[3] = {ws + 1900544, ws + 1933312, ws + 1966080};

  for (int a = 0; a < 3; ++a) {
    prep_wT_bf16<<<1024, 256, 0, stream>>>(Wk[a], wkT[a], 1024, 1024);
    prep_wT_bf16<<<1024, 256, 0, stream>>>(Wr[a], wrT[a], 1024, 1024);
    prep_wT_bf16<<<256,  256, 0, stream>>>(Wm[a], wmT[a], 256, 256);
  }
  prep_wT_bf16<<<256, 256, 0, stream>>>(Wh23, wh23T, 256, 256);
  prep_wT_bf16<<<256, 256, 0, stream>>>(Wc23, wc23T, 256, 256);
  prep_wT_bf16<<<128, 256, 0, stream>>>(Wo[0], woT[0], NC1, NOPAD);
  prep_wT_bf16<<<128, 256, 0, stream>>>(Wo[1], woT[1], NC2, NOPAD);
  prep_wT_bf16<<<128, 256, 0, stream>>>(Wo[2], woT[2], NC3, NOPAD);

  FusedParams prm;
  prm.x = x;
  prm.wkT1 = wkT[0]; prm.wrT1 = wrT[0];
  prm.wkT2 = wkT[1]; prm.wrT2 = wrT[1];
  prm.wkT3 = wkT[2]; prm.wrT3 = wrT[2];
  prm.wh23T = wh23T; prm.wc23T = wc23T;
  prm.wmT1 = wmT[0]; prm.wmT2 = wmT[1]; prm.wmT3 = wmT[2];
  prm.woT1 = woT[0]; prm.woT2 = woT[1]; prm.woT3 = woT[2];
  prm.b1 = b[0]; prm.b2 = b[1]; prm.b3 = b[2];
  prm.bh23 = bh23; prm.bc23 = bc23;
  prm.bm1 = bm[0]; prm.bm2 = bm[1]; prm.bm3 = bm[2];
  prm.bo1 = bo[0]; prm.bo2 = bo[1]; prm.bo3 = bo[2];
  prm.out = (float*)d_out;

  fused_lstm_decoder<<<BATCH / TILE_M, 256, 131072, stream>>>(prm);
}
